// InterpretableMultiHeadAttention_82867099009748
// MI455X (gfx1250) — compile-verified, matrix operand order checked by reading, not run on hardware
//
#include <hip/hip_runtime.h>

typedef __attribute__((ext_vector_type(16))) __bf16 v16bf;
typedef __attribute__((ext_vector_type(8)))  __bf16 v8bf;
typedef __attribute__((ext_vector_type(8)))  float  v8f;
typedef __attribute__((ext_vector_type(4)))  float  v4f;

#define NEG_BIG (-3.402823466e+38f)

// Low 32 bits of a flat-cast LDS pointer == LDS byte offset (ISA 10.2).
__device__ __forceinline__ unsigned lds_off(const void* p) {
    return (unsigned)(size_t)p;
}

// ---------------------------------------------------------------------------
// Generic tiled GEMM: C[M,N] = A[M,Kd](f32) * W[Kd,N](f32) + bias, bf16 WMMA.
// Block: 256 threads = 8 waves, block tile 32(M) x 64(N), K-step 32.
// Output selectable: f32 row-major, bf16 row-major, bf16 transposed [N,ldT].
// ---------------------------------------------------------------------------
#define LDK 40  // padded LDS k-stride (bf16 elems); 80B rows keep 16B alignment

__global__ __launch_bounds__(256) void proj_gemm_kernel(
    const float* __restrict__ A, const float* __restrict__ W,
    const float* __restrict__ bias,
    float* __restrict__ outF, __bf16* __restrict__ outB,
    __bf16* __restrict__ outBT,
    int M, int Kd, int N, int ldT)
{
    __shared__ __align__(16) __bf16 As[32][LDK];
    __shared__ __align__(16) __bf16 Bt[64][LDK];

    const int t    = threadIdx.x;
    const int lane = t & 31, wave = t >> 5;
    const int half = lane >> 4, l16 = lane & 15;
    const int wr   = wave >> 2;      // 0..1  (M sub-tile)
    const int wc   = wave & 3;       // 0..3  (N sub-tile)
    const int m0   = blockIdx.y * 32;
    const int n0   = blockIdx.x * 64;

    v8f acc = {};
    for (int k0 = 0; k0 < Kd; k0 += 32) {
        __syncthreads();
        // Stage A tile 32x32 (f32 -> bf16)
        {
            const int r = t >> 3, c = (t & 7) * 4;
            v4f a4 = *(const v4f*)(A + (size_t)(m0 + r) * Kd + k0 + c);
            As[r][c + 0] = (__bf16)a4[0];
            As[r][c + 1] = (__bf16)a4[1];
            As[r][c + 2] = (__bf16)a4[2];
            As[r][c + 3] = (__bf16)a4[3];
        }
        // Stage W tile 32x64, transposed -> Bt[n][k] so B-fragment reads are rows
        {
            const int kr = t >> 4, c = (t & 15) * 4;
            v4f w0 = *(const v4f*)(W + (size_t)(k0 + kr) * N + n0 + c);
            v4f w1 = *(const v4f*)(W + (size_t)(k0 + kr + 16) * N + n0 + c);
            #pragma unroll
            for (int j = 0; j < 4; ++j) {
                Bt[c + j][kr]      = (__bf16)w0[j];
                Bt[c + j][kr + 16] = (__bf16)w1[j];
            }
        }
        __syncthreads();
        // Fragments per ISA layout: K chunks [8h,8h+8) and [16+8h,16+8h+8)
        v16bf af, bfv;
        const v8bf a0 = *(const v8bf*)&As[wr * 16 + l16][8 * half];
        const v8bf a1 = *(const v8bf*)&As[wr * 16 + l16][16 + 8 * half];
        const v8bf b0 = *(const v8bf*)&Bt[wc * 16 + l16][8 * half];
        const v8bf b1 = *(const v8bf*)&Bt[wc * 16 + l16][16 + 8 * half];
        #pragma unroll
        for (int i = 0; i < 8; ++i) {
            af[i] = a0[i]; af[8 + i] = a1[i];
            bfv[i] = b0[i]; bfv[8 + i] = b1[i];
        }
        acc = __builtin_amdgcn_wmma_f32_16x16x32_bf16(
            false, af, false, bfv, (short)0, acc, false, false);
    }

    const int n = n0 + wc * 16 + l16;
    const float bv = bias ? bias[n] : 0.0f;
    #pragma unroll
    for (int r = 0; r < 8; ++r) {
        const int m = m0 + wr * 16 + r + 8 * half;
        const float c = acc[r] + bv;
        if (outF)  outF[(size_t)m * N + n]   = c;
        if (outB)  outB[(size_t)m * N + n]   = (__bf16)c;
        if (outBT) outBT[(size_t)n * ldT + m] = (__bf16)c;
    }
}

// ---------------------------------------------------------------------------
// Scores: attn[b,h,q,k] = (Q_bh[q,:] . K_bh[k,:]) / 8, masked.
// Block 256 = 8 waves covering 16(q) x 128(k) for one (b,h).
// Q (16x64) and K (128x64) bf16 tiles staged into LDS with ASYNC loads
// (GLOBAL_LOAD_ASYNC_TO_LDS_B128, ASYNCcnt), then ds_load_b128 fragments.
// ---------------------------------------------------------------------------
__global__ __launch_bounds__(256) void scores_kernel(
    const __bf16* __restrict__ Qp, const __bf16* __restrict__ Kp,
    const unsigned char* __restrict__ mask, float* __restrict__ attn)
{
    __shared__ __align__(16) __bf16 Qs[16][64];    //  2 KB
    __shared__ __align__(16) __bf16 Ks[128][64];   // 16 KB

    const int t    = threadIdx.x;
    const int lane = t & 31, wave = t >> 5;
    const int half = lane >> 4, l16 = lane & 15;
    const int bh = blockIdx.z, b = bh >> 4, h = bh & 15;
    const int q0  = blockIdx.y * 16;
    const int kt0 = blockIdx.x * 128;

    // ---- async stage Q tile: 16 rows x 64 bf16 = 128 chunks of 16B ----
    if (t < 128) {
        const int row = t >> 3, c = (t & 7) * 8;
        const __bf16* g = Qp + (size_t)(b * 2048 + q0 + row) * 1024 + h * 64 + c;
        const unsigned l = lds_off(&Qs[row][c]);
        asm volatile("global_load_async_to_lds_b128 %0, %1, off"
                     :: "v"(l), "v"(g) : "memory");
    }
    // ---- async stage K tile: 128 rows x 64 bf16 = 1024 chunks, 4/thread ----
    #pragma unroll
    for (int i = 0; i < 4; ++i) {
        const int c16 = t + 256 * i;
        const int row = c16 >> 3, c = (c16 & 7) * 8;
        const __bf16* g = Kp + (size_t)(b * 2048 + kt0 + row) * 1024 + h * 64 + c;
        const unsigned l = lds_off(&Ks[row][c]);
        asm volatile("global_load_async_to_lds_b128 %0, %1, off"
                     :: "v"(l), "v"(g) : "memory");
    }
    asm volatile("s_wait_asynccnt 0x0" ::: "memory");
    __syncthreads();

    // ---- per-wave 16x16 tile from LDS fragments ----
    v8f acc = {};
    #pragma unroll
    for (int ks = 0; ks < 64; ks += 32) {
        v16bf af, bfv;
        const v8bf a0 = *(const v8bf*)&Qs[l16][ks + 8 * half];
        const v8bf a1 = *(const v8bf*)&Qs[l16][ks + 16 + 8 * half];
        const v8bf b0 = *(const v8bf*)&Ks[wave * 16 + l16][ks + 8 * half];
        const v8bf b1 = *(const v8bf*)&Ks[wave * 16 + l16][ks + 16 + 8 * half];
        #pragma unroll
        for (int i = 0; i < 8; ++i) {
            af[i] = a0[i]; af[8 + i] = a1[i];
            bfv[i] = b0[i]; bfv[8 + i] = b1[i];
        }
        acc = __builtin_amdgcn_wmma_f32_16x16x32_bf16(
            false, af, false, bfv, (short)0, acc, false, false);
    }

    const int kidx = kt0 + wave * 16 + l16;
    #pragma unroll
    for (int r = 0; r < 8; ++r) {
        const int q = q0 + r + 8 * half;
        float s = acc[r] * 0.125f;  // 1/sqrt(64)
        const size_t mi = (size_t)b * (2048ull * 2048) + (size_t)q * 2048 + kidx;
        if (!mask[mi]) s = NEG_BIG;
        attn[(size_t)bh * (2048ull * 2048) + (size_t)q * 2048 + kidx] = s;
    }
}

// ---------------------------------------------------------------------------
// Row softmax over attn (in place). One 256-thread block per row of 2048.
// ---------------------------------------------------------------------------
__global__ __launch_bounds__(256) void softmax_kernel(float* __restrict__ attn)
{
    __shared__ float red[256];
    float* p = attn + (size_t)blockIdx.x * 2048;
    const int t = threadIdx.x;

    float vals[8];
    float m = NEG_BIG;
    #pragma unroll
    for (int i = 0; i < 8; ++i) { vals[i] = p[t + i * 256]; m = fmaxf(m, vals[i]); }
    red[t] = m; __syncthreads();
    for (int s = 128; s > 0; s >>= 1) {
        if (t < s) red[t] = fmaxf(red[t], red[t + s]);
        __syncthreads();
    }
    m = red[0]; __syncthreads();

    float sum = 0.0f;
    #pragma unroll
    for (int i = 0; i < 8; ++i) { vals[i] = __expf(vals[i] - m); sum += vals[i]; }
    red[t] = sum; __syncthreads();
    for (int s = 128; s > 0; s >>= 1) {
        if (t < s) red[t] += red[t + s];
        __syncthreads();
    }
    const float inv = 1.0f / red[0];
    #pragma unroll
    for (int i = 0; i < 8; ++i) p[t + i * 256] = vals[i] * inv;
}

// ---------------------------------------------------------------------------
// attn[b,h] (2048x2048, f32) @ Vh[b] (2048x64) -> Ohead[b,h] (2048x64, f32).
// Block 128 = 4 waves; wave w -> d-tile w (16 cols); 16q x 64d per block.
// A fragments: f32 attn converted to bf16 on load; B: contiguous rows of VhT.
// ---------------------------------------------------------------------------
__global__ __launch_bounds__(128) void av_kernel(
    const float* __restrict__ attn, const __bf16* __restrict__ VhT,
    float* __restrict__ Ohead)
{
    const int t    = threadIdx.x;
    const int lane = t & 31, wave = t >> 5;
    const int half = lane >> 4, l16 = lane & 15;
    const int bh = blockIdx.y, b = bh >> 4;
    const int q0 = blockIdx.x * 16;
    const int n  = wave * 16 + l16;            // 0..63 (d_head)

    const float*  arow = attn + (size_t)bh * (2048ull * 2048) + (size_t)(q0 + l16) * 2048;
    const __bf16* vrow = VhT + (size_t)n * 8192 + b * 2048;

    v8f acc = {};
    for (int tk = 0; tk < 2048; tk += 32) {
        __builtin_prefetch(arow + tk + 256, 0, 0);   // global_prefetch_b8
        v16bf af, bfv;
        const v4f a00 = *(const v4f*)(arow + tk + 8 * half);
        const v4f a01 = *(const v4f*)(arow + tk + 8 * half + 4);
        const v4f a10 = *(const v4f*)(arow + tk + 16 + 8 * half);
        const v4f a11 = *(const v4f*)(arow + tk + 16 + 8 * half + 4);
        const v8bf b0 = *(const v8bf*)(vrow + tk + 8 * half);
        const v8bf b1 = *(const v8bf*)(vrow + tk + 16 + 8 * half);
        #pragma unroll
        for (int i = 0; i < 4; ++i) {
            af[i]      = (__bf16)a00[i];
            af[4 + i]  = (__bf16)a01[i];
            af[8 + i]  = (__bf16)a10[i];
            af[12 + i] = (__bf16)a11[i];
        }
        #pragma unroll
        for (int i = 0; i < 8; ++i) { bfv[i] = b0[i]; bfv[8 + i] = b1[i]; }
        acc = __builtin_amdgcn_wmma_f32_16x16x32_bf16(
            false, af, false, bfv, (short)0, acc, false, false);
    }

    #pragma unroll
    for (int r = 0; r < 8; ++r) {
        const int q = q0 + r + 8 * half;
        Ohead[((size_t)bh * 2048 + q) * 64 + n] = acc[r];
    }
}

// ---------------------------------------------------------------------------
// Omean[b,q,d] = (1/16) * sum_h Ohead[b,h,q,d]
// ---------------------------------------------------------------------------
__global__ __launch_bounds__(256) void head_mean_kernel(
    const float* __restrict__ Ohead, float* __restrict__ Omean)
{
    const size_t idx = (size_t)blockIdx.x * blockDim.x + threadIdx.x; // 8192*64
    const int m = (int)(idx >> 6);     // token 0..8191
    const int d = (int)(idx & 63);
    const int b = m >> 11, tq = m & 2047;
    float s = 0.0f;
    #pragma unroll
    for (int h = 0; h < 16; ++h)
        s += Ohead[(((size_t)(b * 16 + h) * 2048) + tq) * 64 + d];
    Omean[idx] = s * (1.0f / 16.0f);
}

// ---------------------------------------------------------------------------
extern "C" void kernel_launch(void* const* d_in, const int* in_sizes, int n_in,
                              void* d_out, int out_size, void* d_ws, size_t ws_size,
                              hipStream_t stream) {
    const float* q  = (const float*)d_in[0];
    const float* k  = (const float*)d_in[1];
    const float* v  = (const float*)d_in[2];
    const unsigned char* mask = (const unsigned char*)d_in[3];
    const float* Wq = (const float*)d_in[4];
    const float* bq = (const float*)d_in[5];
    const float* Wk = (const float*)d_in[6];
    const float* bk = (const float*)d_in[7];
    const float* Wv = (const float*)d_in[8];
    const float* bv = (const float*)d_in[9];
    const float* Wo = (const float*)d_in[10];
    const float* bo = (const float*)d_in[11];

    float* out  = (float*)d_out;                    // [4,2048,1024]
    float* attn = out + (size_t)4 * 2048 * 1024;    // [4,16,2048,2048]

    char* ws = (char*)d_ws;
    __bf16* Qp    = (__bf16*)ws; ws += (size_t)8192 * 1024 * 2;   // 16 MB
    __bf16* Kp    = (__bf16*)ws; ws += (size_t)8192 * 1024 * 2;   // 16 MB
    __bf16* VhT   = (__bf16*)ws; ws += (size_t)64 * 8192 * 2;     //  1 MB
    float*  Ohead = (float*)ws;  ws += (size_t)64 * 2048 * 64 * 4; // 32 MB
    float*  Omean = (float*)ws;  ws += (size_t)8192 * 64 * 4;      //  2 MB

    const dim3 blk(256);
    // Q = q@Wq + bq  -> bf16 [8192,1024]
    proj_gemm_kernel<<<dim3(16, 256), blk, 0, stream>>>(
        q, Wq, bq, nullptr, Qp, nullptr, 8192, 1024, 1024, 0);
    // K = k@Wk + bk  -> bf16 [8192,1024]
    proj_gemm_kernel<<<dim3(16, 256), blk, 0, stream>>>(
        k, Wk, bk, nullptr, Kp, nullptr, 8192, 1024, 1024, 0);
    // Vh = v@Wv + bv -> bf16 transposed [64,8192]
    proj_gemm_kernel<<<dim3(1, 256), blk, 0, stream>>>(
        v, Wv, bv, nullptr, nullptr, VhT, 8192, 1024, 64, 8192);
    // scores -> attn (f32, masked, scaled), async-LDS staged
    scores_kernel<<<dim3(16, 128, 64), blk, 0, stream>>>(Qp, Kp, mask, attn);
    // softmax rows in place
    softmax_kernel<<<dim3(131072), blk, 0, stream>>>(attn);
    // per-head attn @ Vh
    av_kernel<<<dim3(128, 64), dim3(128), 0, stream>>>(attn, VhT, Ohead);
    // mean over heads
    head_mean_kernel<<<dim3(2048), blk, 0, stream>>>(Ohead, Omean);
    // out = Omean @ Wo + bo -> f32 [8192,1024]
    proj_gemm_kernel<<<dim3(16, 256), blk, 0, stream>>>(
        Omean, Wo, bo, out, nullptr, nullptr, 8192, 64, 1024, 0);
}
